// MultiHeadAttention_41463614275821
// MI455X (gfx1250) — matrix operand order repaired and checked by reading, not compiled, not run on hardware
//
#include <hip/hip_runtime.h>
#include <hip/hip_bf16.h>
#include <math.h>

// ---------------------------------------------------------------------------
// MHA for MI455X (gfx1250): bf16 WMMA GEMMs + f32 softmax/LayerNorm.
//  B=4, S=2048, D_MODEL=512, H=8, DK=DV=64, SCALE=1/sqrt(1e-5).
//  Block tile 256x64 (8 waves x 32 rows), double-buffered B tile in LDS,
//  A fragments fed directly from global memory (2x b128/lane).
// ---------------------------------------------------------------------------

typedef __attribute__((ext_vector_type(16))) __bf16 v16bf;
typedef __attribute__((ext_vector_type(8)))  __bf16 v8bf;
typedef __attribute__((ext_vector_type(8)))  float  v8f;

#define BB   4
#define SS   2048
#define DM   512
#define NH   8
#define DK   64
#define BLD  40                    // B-tile LDS pitch in bf16 elems (80 B)
#define MHA_SCALE 316.22776601683796f
#define LNEPS 1e-5f

// --------------------------- WMMA helpers ----------------------------------

__device__ __forceinline__ v8f wmma_bf16(v16bf a, v16bf b, v8f c) {
  return __builtin_amdgcn_wmma_f32_16x16x32_bf16(false, a, false, b, (short)0, c,
                                                 false, false);
}

__device__ __forceinline__ v16bf combine16(v8bf lo, v8bf hi) {
  v16bf a;
#pragma unroll
  for (int e = 0; e < 8; ++e) { a[e] = lo[e]; a[e + 8] = hi[e]; }
  return a;
}

// A-frag (16x32 bf16, ISA 7.12.2): lane reads its own M-row; K runs
// [half*8 .. +7] and [16+half*8 .. +7] are contiguous in memory.
__device__ __forceinline__ v16bf load_a_frag_bf16(const __bf16* row_k0,
                                                  int lane) {
  const __bf16* p = row_k0 + ((lane >> 4) << 3);
  v8bf lo = *(const v8bf*)(p);          // global_load_b128
  v8bf hi = *(const v8bf*)(p + 16);     // global_load_b128
  return combine16(lo, hi);
}

__device__ __forceinline__ v16bf load_a_frag_f32(const float* row_k0,
                                                 int lane) {
  const float* p = row_k0 + ((lane >> 4) << 3);
  v8f lo = *(const v8f*)(p);            // 2x global_load_b128
  v8f hi = *(const v8f*)(p + 16);
  v16bf a;
#pragma unroll
  for (int e = 0; e < 8; ++e) {
    a[e]     = (__bf16)lo[e];
    a[e + 8] = (__bf16)hi[e];
  }
  return a;
}

// B-frag (32x16 bf16) from LDS tile stored transposed: Bt[n][kk], pitch BLD.
// Lane (half,n) reads Bt[col0+n][half*8 .. +7] and [16+half*8 .. +7] -> 2x ds_load_b128.
__device__ __forceinline__ v16bf load_b_frag(const __bf16* Bt, int col0,
                                             int lane) {
  const __bf16* p = Bt + (size_t)(col0 + (lane & 15)) * BLD + ((lane >> 4) << 3);
  v8bf lo = *(const v8bf*)(p);
  v8bf hi = *(const v8bf*)(p + 16);
  return combine16(lo, hi);
}

// ----------------------- B-tile fill helpers (32xK -> Bt[n][kk]) ------------

// From row-major f32 weights W[kbase+r][n0+c]: vector row loads, scatter b16.
__device__ __forceinline__ void fillB_w_f32(__bf16* Bt, const float* W,
                                            int kbase, int n0, int tid) {
  const int r = tid >> 3, cg = (tid & 7) * 8;
  v8f wv = *(const v8f*)(W + (size_t)(kbase + r) * DM + n0 + cg);
#pragma unroll
  for (int j = 0; j < 8; ++j) Bt[(size_t)(cg + j) * BLD + r] = (__bf16)wv[j];
}

// From k rows (B = k^T): contiguous bf16 in == contiguous bf16 out (b128/b128).
__device__ __forceinline__ void fillB_kT(__bf16* Bt, const __bf16* kmat,
                                         int kbase, int n0, int tid) {
  const int n = tid >> 2, kg = (tid & 3) * 8;
  v8bf vv = *(const v8bf*)(kmat + (size_t)(n0 + n) * DK + kbase + kg);
  *(v8bf*)(Bt + (size_t)n * BLD + kg) = vv;           // ds_store_b128
}

// From v rows (row-major bf16 [K][64]): vector row loads, scatter b16.
__device__ __forceinline__ void fillB_vrows(__bf16* Bt, const __bf16* V,
                                            int kbase, int tid) {
  const int kk = tid >> 3, ng = (tid & 7) * 8;
  v8bf vv = *(const v8bf*)(V + (size_t)(kbase + kk) * DK + ng);
#pragma unroll
  for (int j = 0; j < 8; ++j) Bt[(size_t)(ng + j) * BLD + kk] = vv[j];
}

// ---------------------------------------------------------------------------
// Kernel 1: fused QKV projection.  X[8192,512]f32 @ W[512,512]f32 + bias
//   -> bf16 [B,H,S,64].  blockIdx.z selects q/k/v.  Tile 256x64.
// ---------------------------------------------------------------------------
__global__ __launch_bounds__(256) void qkv_proj_kernel(
    const float* __restrict__ Qin, const float* __restrict__ Kin,
    const float* __restrict__ Vin,
    const float* __restrict__ Wq, const float* __restrict__ bq,
    const float* __restrict__ Wk, const float* __restrict__ bk,
    const float* __restrict__ Wv, const float* __restrict__ bv,
    __bf16* __restrict__ qb, __bf16* __restrict__ kb, __bf16* __restrict__ vb) {
  const int which = blockIdx.z;
  const float* X    = (which == 0) ? Qin : (which == 1) ? Kin : Vin;
  const float* W    = (which == 0) ? Wq  : (which == 1) ? Wk  : Wv;
  const float* bias = (which == 0) ? bq  : (which == 1) ? bk  : bv;
  __bf16* out       = (which == 0) ? qb  : (which == 1) ? kb  : vb;

  __shared__ __bf16 Bt[2][64 * BLD];

  const int tid  = threadIdx.x;
  const int lane = tid & 31;
  const int wave = tid >> 5;
  const int m0   = blockIdx.x * 256;
  const int n0   = blockIdx.y * 64;

  const float* Arow0 = X + (size_t)(m0 + wave * 32 + (lane & 15)) * DM;
  const float* Arow1 = Arow0 + (size_t)16 * DM;

  const v8f z = {0.f, 0.f, 0.f, 0.f, 0.f, 0.f, 0.f, 0.f};
  v8f acc0[4] = {z, z, z, z};
  v8f acc1[4] = {z, z, z, z};

  const int NK = DM / 32;
  fillB_w_f32(Bt[0], W, 0, n0, tid);
  for (int ks = 0; ks < NK; ++ks) {
    __syncthreads();
    if (ks + 1 < NK) fillB_w_f32(Bt[(ks + 1) & 1], W, (ks + 1) * 32, n0, tid);
    const __bf16* buf = Bt[ks & 1];
    const int kbase = ks * 32;
    __builtin_prefetch(Arow0 + kbase + 32, 0, 1);      // global_prefetch_b8
    const v16bf a0 = load_a_frag_f32(Arow0 + kbase, lane);
    const v16bf a1 = load_a_frag_f32(Arow1 + kbase, lane);
#pragma unroll
    for (int t = 0; t < 4; ++t) {
      const v16bf bfr = load_b_frag(buf, t * 16, lane);
      acc0[t] = wmma_bf16(a0, bfr, acc0[t]);
      acc1[t] = wmma_bf16(a1, bfr, acc1[t]);
    }
  }

  const int half = lane >> 4, nn = lane & 15;
#pragma unroll
  for (int t = 0; t < 4; ++t) {
    const int ng = n0 + t * 16 + nn;                    // -> (head, d)
    const int h = ng >> 6, d = ng & 63;
    const float bia = bias[ng];
#pragma unroll
    for (int p = 0; p < 2; ++p) {
      const v8f* acc = p ? acc1 : acc0;
#pragma unroll
      for (int r = 0; r < 8; ++r) {
        const int mg = m0 + wave * 32 + p * 16 + half * 8 + r;   // -> (b, s)
        const int bi = mg >> 11, s = mg & (SS - 1);
        out[(((size_t)bi * NH + h) * SS + s) * DK + d] =
            (__bf16)(acc[t][r] + bia);
      }
    }
  }
}

// ---------------------------------------------------------------------------
// Kernel 2: scores = (q @ k^T) * SCALE  per (b,h).  M=N=2048, K=64.
// ---------------------------------------------------------------------------
__global__ __launch_bounds__(256) void scores_kernel(
    const __bf16* __restrict__ qb, const __bf16* __restrict__ kbuf,
    float* __restrict__ attn) {
  const int bh = blockIdx.z;
  const __bf16* q = qb   + (size_t)bh * SS * DK;
  const __bf16* k = kbuf + (size_t)bh * SS * DK;
  float* out      = attn + (size_t)bh * SS * SS;

  __shared__ __bf16 Bt[2][64 * BLD];

  const int tid = threadIdx.x, lane = tid & 31, wave = tid >> 5;
  const int m0 = blockIdx.x * 256, n0 = blockIdx.y * 64;

  const __bf16* Arow0 = q + (size_t)(m0 + wave * 32 + (lane & 15)) * DK;
  const __bf16* Arow1 = Arow0 + (size_t)16 * DK;

  const v8f z = {0.f, 0.f, 0.f, 0.f, 0.f, 0.f, 0.f, 0.f};
  v8f acc0[4] = {z, z, z, z};
  v8f acc1[4] = {z, z, z, z};

  const int NK = DK / 32;
  fillB_kT(Bt[0], k, 0, n0, tid);
  for (int ks = 0; ks < NK; ++ks) {
    __syncthreads();
    if (ks + 1 < NK) fillB_kT(Bt[(ks + 1) & 1], k, (ks + 1) * 32, n0, tid);
    const __bf16* buf = Bt[ks & 1];
    const int kbase = ks * 32;
    const v16bf a0 = load_a_frag_bf16(Arow0 + kbase, lane);
    const v16bf a1 = load_a_frag_bf16(Arow1 + kbase, lane);
#pragma unroll
    for (int t = 0; t < 4; ++t) {
      const v16bf bfr = load_b_frag(buf, t * 16, lane);
      acc0[t] = wmma_bf16(a0, bfr, acc0[t]);
      acc1[t] = wmma_bf16(a1, bfr, acc1[t]);
    }
  }

  const int half = lane >> 4, nn = lane & 15;
#pragma unroll
  for (int t = 0; t < 4; ++t) {
#pragma unroll
    for (int p = 0; p < 2; ++p) {
      const v8f* acc = p ? acc1 : acc0;
#pragma unroll
      for (int r = 0; r < 8; ++r) {
        const int mg = m0 + wave * 32 + p * 16 + half * 8 + r;
        out[(size_t)mg * SS + n0 + t * 16 + nn] = acc[t][r] * MHA_SCALE;
      }
    }
  }
}

// ---------------------------------------------------------------------------
// Kernel 3: row softmax in place. One block per row of 2048.
// ---------------------------------------------------------------------------
__global__ __launch_bounds__(256) void softmax_kernel(float* __restrict__ attn) {
  float* p = attn + (size_t)blockIdx.x * SS;
  const int tid = threadIdx.x, lane = tid & 31, wave = tid >> 5;
  __shared__ float red[8];

  float v[8];
  float mx = -INFINITY;
#pragma unroll
  for (int i = 0; i < 8; ++i) {
    v[i] = p[tid + i * 256];
    mx = fmaxf(mx, v[i]);
  }
#pragma unroll
  for (int o = 16; o > 0; o >>= 1) mx = fmaxf(mx, __shfl_xor(mx, o, 32));
  if (lane == 0) red[wave] = mx;
  __syncthreads();
  float bm = red[0];
#pragma unroll
  for (int w = 1; w < 8; ++w) bm = fmaxf(bm, red[w]);
  __syncthreads();

  float sum = 0.f;
#pragma unroll
  for (int i = 0; i < 8; ++i) {
    v[i] = __expf(v[i] - bm);
    sum += v[i];
  }
#pragma unroll
  for (int o = 16; o > 0; o >>= 1) sum += __shfl_xor(sum, o, 32);
  if (lane == 0) red[wave] = sum;
  __syncthreads();
  float bs = 0.f;
#pragma unroll
  for (int w = 0; w < 8; ++w) bs += red[w];
  const float inv = 1.0f / bs;
#pragma unroll
  for (int i = 0; i < 8; ++i) p[tid + i * 256] = v[i] * inv;
}

// ---------------------------------------------------------------------------
// Kernel 4: context = attn @ v per (b,h).  M=2048, N=64, K=2048.
//   ctx stored bf16 as [B, S, H*64].
// ---------------------------------------------------------------------------
__global__ __launch_bounds__(256) void context_kernel(
    const float* __restrict__ attn, const __bf16* __restrict__ vbuf,
    __bf16* __restrict__ ctx) {
  const int bh = blockIdx.z;
  const int b = bh >> 3, h = bh & 7;
  const float*  A = attn + (size_t)bh * SS * SS;
  const __bf16* V = vbuf + (size_t)bh * SS * DK;

  __shared__ __bf16 Bt[2][64 * BLD];

  const int tid = threadIdx.x, lane = tid & 31, wave = tid >> 5;
  const int m0 = blockIdx.x * 256;

  const float* Arow0 = A + (size_t)(m0 + wave * 32 + (lane & 15)) * SS;
  const float* Arow1 = Arow0 + (size_t)16 * SS;

  const v8f z = {0.f, 0.f, 0.f, 0.f, 0.f, 0.f, 0.f, 0.f};
  v8f acc0[4] = {z, z, z, z};
  v8f acc1[4] = {z, z, z, z};

  const int NK = SS / 32;
  fillB_vrows(Bt[0], V, 0, tid);
  for (int ks = 0; ks < NK; ++ks) {
    __syncthreads();
    if (ks + 1 < NK) fillB_vrows(Bt[(ks + 1) & 1], V, (ks + 1) * 32, tid);
    const __bf16* buf = Bt[ks & 1];
    const int kbase = ks * 32;
    __builtin_prefetch(Arow0 + kbase + 32, 0, 1);      // global_prefetch_b8
    const v16bf a0 = load_a_frag_f32(Arow0 + kbase, lane);
    const v16bf a1 = load_a_frag_f32(Arow1 + kbase, lane);
#pragma unroll
    for (int t = 0; t < 4; ++t) {
      const v16bf bfr = load_b_frag(buf, t * 16, lane);
      acc0[t] = wmma_bf16(a0, bfr, acc0[t]);
      acc1[t] = wmma_bf16(a1, bfr, acc1[t]);
    }
  }

  const int half = lane >> 4, nn = lane & 15;
#pragma unroll
  for (int t = 0; t < 4; ++t) {
    const int d = t * 16 + nn;
#pragma unroll
    for (int p = 0; p < 2; ++p) {
      const v8f* acc = p ? acc1 : acc0;
#pragma unroll
      for (int r = 0; r < 8; ++r) {
        const int s = m0 + wave * 32 + p * 16 + half * 8 + r;
        ctx[((size_t)b * SS + s) * DM + h * DK + d] = (__bf16)acc[t][r];
      }
    }
  }
}

// ---------------------------------------------------------------------------
// Kernel 5: out-proj + bias + residual.  ctx[8192,512]bf16 @ Wo[512,512]f32
//   -> proj f32 (pre-LayerNorm).
// ---------------------------------------------------------------------------
__global__ __launch_bounds__(256) void outproj_kernel(
    const __bf16* __restrict__ ctx, const float* __restrict__ Wo,
    const float* __restrict__ bo, const float* __restrict__ Qin,
    float* __restrict__ proj) {
  __shared__ __bf16 Bt[2][64 * BLD];

  const int tid = threadIdx.x, lane = tid & 31, wave = tid >> 5;
  const int m0 = blockIdx.x * 256, n0 = blockIdx.y * 64;

  const __bf16* Arow0 = ctx + (size_t)(m0 + wave * 32 + (lane & 15)) * DM;
  const __bf16* Arow1 = Arow0 + (size_t)16 * DM;

  const v8f z = {0.f, 0.f, 0.f, 0.f, 0.f, 0.f, 0.f, 0.f};
  v8f acc0[4] = {z, z, z, z};
  v8f acc1[4] = {z, z, z, z};

  const int NK = DM / 32;
  fillB_w_f32(Bt[0], Wo, 0, n0, tid);
  for (int ks = 0; ks < NK; ++ks) {
    __syncthreads();
    if (ks + 1 < NK) fillB_w_f32(Bt[(ks + 1) & 1], Wo, (ks + 1) * 32, n0, tid);
    const __bf16* buf = Bt[ks & 1];
    const int kbase = ks * 32;
    __builtin_prefetch(Arow0 + kbase + 32, 0, 1);
    const v16bf a0 = load_a_frag_bf16(Arow0 + kbase, lane);
    const v16bf a1 = load_a_frag_bf16(Arow1 + kbase, lane);
#pragma unroll
    for (int t = 0; t < 4; ++t) {
      const v16bf bfr = load_b_frag(buf, t * 16, lane);
      acc0[t] = wmma_bf16(a0, bfr, acc0[t]);
      acc1[t] = wmma_bf16(a1, bfr, acc1[t]);
    }
  }

  const int half = lane >> 4, nn = lane & 15;
#pragma unroll
  for (int t = 0; t < 4; ++t) {
    const int ng = n0 + t * 16 + nn;
    const float bia = bo[ng];
#pragma unroll
    for (int p = 0; p < 2; ++p) {
      const v8f* acc = p ? acc1 : acc0;
#pragma unroll
      for (int r = 0; r < 8; ++r) {
        const int mg = m0 + wave * 32 + p * 16 + half * 8 + r;
        proj[(size_t)mg * DM + ng] =
            acc[t][r] + bia + Qin[(size_t)mg * DM + ng];   // residual
      }
    }
  }
}

// ---------------------------------------------------------------------------
// Kernel 6: LayerNorm per row (512 features). One block per row.
// ---------------------------------------------------------------------------
__global__ __launch_bounds__(256) void layernorm_kernel(
    const float* __restrict__ proj, const float* __restrict__ gamma,
    const float* __restrict__ beta, float* __restrict__ out) {
  const size_t row = blockIdx.x;
  const float* p = proj + row * DM;
  const int tid = threadIdx.x, lane = tid & 31, wave = tid >> 5;
  __shared__ float rsum[8], rsq[8];

  const float x0 = p[tid];
  const float x1 = p[tid + 256];
  float s  = x0 + x1;
  float s2 = x0 * x0 + x1 * x1;
#pragma unroll
  for (int o = 16; o > 0; o >>= 1) {
    s  += __shfl_xor(s, o, 32);
    s2 += __shfl_xor(s2, o, 32);
  }
  if (lane == 0) { rsum[wave] = s; rsq[wave] = s2; }
  __syncthreads();
  float ts = 0.f, ts2 = 0.f;
#pragma unroll
  for (int w = 0; w < 8; ++w) { ts += rsum[w]; ts2 += rsq[w]; }
  const float mu  = ts * (1.0f / DM);
  const float var = ts2 * (1.0f / DM) - mu * mu;
  const float ri  = rsqrtf(var + LNEPS);

  out[row * DM + tid]       = (x0 - mu) * ri * gamma[tid]       + beta[tid];
  out[row * DM + tid + 256] = (x1 - mu) * ri * gamma[tid + 256] + beta[tid + 256];
}

// ---------------------------------------------------------------------------
// Host launcher.
// ---------------------------------------------------------------------------
extern "C" void kernel_launch(void* const* d_in, const int* in_sizes, int n_in,
                              void* d_out, int out_size, void* d_ws,
                              size_t ws_size, hipStream_t stream) {
  (void)in_sizes; (void)n_in; (void)out_size; (void)ws_size;

  const float* Qin   = (const float*)d_in[0];
  const float* Kin   = (const float*)d_in[1];
  const float* Vin   = (const float*)d_in[2];
  const float* Wq    = (const float*)d_in[3];
  const float* bq    = (const float*)d_in[4];
  const float* Wk    = (const float*)d_in[5];
  const float* bk    = (const float*)d_in[6];
  const float* Wv    = (const float*)d_in[7];
  const float* bv    = (const float*)d_in[8];
  const float* Wo    = (const float*)d_in[9];
  const float* bo    = (const float*)d_in[10];
  const float* gamma = (const float*)d_in[11];
  const float* beta  = (const float*)d_in[12];

  float* out  = (float*)d_out;                       // [4,2048,512]
  float* attn = out + (size_t)BB * SS * DM;          // [4,8,2048,2048]

  char* ws = (char*)d_ws;                            // 48 MiB used
  __bf16* qb  = (__bf16*)(ws);                       //  8 MiB
  __bf16* kb  = (__bf16*)(ws + ((size_t)8  << 20));  //  8 MiB
  __bf16* vb  = (__bf16*)(ws + ((size_t)16 << 20));  //  8 MiB
  __bf16* ctx = (__bf16*)(ws + ((size_t)24 << 20));  //  8 MiB
  float*  prj = (float*) (ws + ((size_t)32 << 20));  // 16 MiB

  qkv_proj_kernel<<<dim3(32, 8, 3), 256, 0, stream>>>(
      Qin, Kin, Vin, Wq, bq, Wk, bk, Wv, bv, qb, kb, vb);
  scores_kernel<<<dim3(8, 32, 32), 256, 0, stream>>>(qb, kb, attn);
  softmax_kernel<<<dim3(BB * NH * SS, 1, 1), 256, 0, stream>>>(attn);
  context_kernel<<<dim3(8, 1, 32), 256, 0, stream>>>(attn, vb, ctx);
  outproj_kernel<<<dim3(32, 8, 1), 256, 0, stream>>>(ctx, Wo, bo, Qin, prj);
  layernorm_kernel<<<dim3(BB * SS, 1, 1), 256, 0, stream>>>(prj, gamma, beta, out);
}
